// RNNModel_10711648436424
// MI455X (gfx1250) — matrix operand order repaired and checked by reading, not compiled, not run on hardware
//
#include <hip/hip_runtime.h>

#ifndef __has_builtin
#define __has_builtin(x) 0
#endif

#if __has_builtin(__builtin_amdgcn_tensor_load_to_lds) && __has_builtin(__builtin_amdgcn_s_wait_tensorcnt)
#define USE_TDM 1
#else
#define USE_TDM 0
#endif

#if __has_builtin(__builtin_amdgcn_tanhf)
#define TANHF(x) __builtin_amdgcn_tanhf(x)
#else
#define TANHF(x) (1.0f - 2.0f / (__expf(2.0f * (x)) + 1.0f))
#endif

typedef float        f32x8   __attribute__((ext_vector_type(8)));
typedef __bf16       bf16x16 __attribute__((ext_vector_type(16)));
typedef __bf16       bf16x8  __attribute__((ext_vector_type(8)));
typedef unsigned int u32x4   __attribute__((ext_vector_type(4)));
typedef int          i32x8   __attribute__((ext_vector_type(8)));
typedef int          i32x4   __attribute__((ext_vector_type(4)));

namespace {
constexpr int SEQT  = 50;
constexpr int BATCH = 8192;
constexpr int HIDC  = 75;
constexpr int LK    = 104;              // padded row stride (bf16 elements) = 208B = 13*16B
constexpr int KCH   = 3;                // K chunks of 32 -> covers K = 0..95
constexpr int MROWS = 32;               // batch rows per block (2 M-tiles)
constexpr int NWAVES = 5;               // one wave per 16-wide N tile (N padded to 80)
constexpr int NTHR  = NWAVES * 32;      // 160 threads
constexpr int TILE_CHUNKS = MROWS * LK * 2 / 16;  // 416 x 16B = 6656B per x tile
}

static __device__ __forceinline__ unsigned short f2bf(float f) {
  unsigned u = __float_as_uint(f);
  unsigned r = u + 0x7FFFu + ((u >> 16) & 1u);   // round-to-nearest-even
  return (unsigned short)(r >> 16);
}

static __device__ __forceinline__ bf16x16 ld16(const unsigned short* p0,
                                               const unsigned short* p1) {
  bf16x8 a = *(const bf16x8*)p0;
  bf16x8 b = *(const bf16x8*)p1;
  return __builtin_shufflevector(a, b, 0, 1, 2, 3, 4, 5, 6, 7,
                                       8, 9, 10, 11, 12, 13, 14, 15);
}

static __device__ __forceinline__ f32x8 wmma_bf16(bf16x16 a, bf16x16 b, f32x8 c) {
  return __builtin_amdgcn_wmma_f32_16x16x32_bf16(false, a, false, b,
                                                 (short)0, c, false, false);
}

#if USE_TDM
// 1-D TDM copy of MROWS*LK bf16 elements (contiguous 6656B) global -> LDS.
// D# layout per CDNA5 ISA ch.8: group0 = {flags, lds_addr, gaddr_lo, gaddr_hi|type},
// group1 carries data_size / tensor_dim0 / tile_dim0 / stride0.
// This toolchain exposes the 6-arg builtin: (g0, g1, g2, g3, g4, cpol).
static __device__ __forceinline__ void tdm_tile_load(const unsigned short* gsrc,
                                                     void* ldst) {
  unsigned long long ga = (unsigned long long)(size_t)gsrc;
  unsigned int lo = (unsigned int)(size_t)ldst;  // LDS byte offset (low 32 bits)
  constexpr unsigned ELEMS = (unsigned)(MROWS * LK);  // 3328 2-byte elements
  u32x4 g0;
  g0[0] = 1u;                                            // count=1, user descriptor
  g0[1] = lo;                                            // lds_addr
  g0[2] = (unsigned int)(ga & 0xFFFFFFFFull);            // global_addr[31:0]
  g0[3] = (unsigned int)((ga >> 32) & 0x01FFFFFFull)     // global_addr[56:32]
        | (2u << 30);                                    // type = 2 ("image")
  i32x8 g1;
  g1[0] = (int)(1u << 16);                               // data_size=1 (2B); wg_mask=0
  g1[1] = (int)((ELEMS & 0xFFFFu) << 16);                // tensor_dim0[15:0]
  g1[2] = (int)((ELEMS >> 16) | (1u << 16));             // tensor_dim0[31:16]; tensor_dim1=1
  g1[3] = (int)((ELEMS & 0xFFFFu) << 16);                // tile_dim0 = 3328
  g1[4] = 0;                                             // tile_dim1=0, tile_dim2=0
  g1[5] = (int)ELEMS;                                    // tensor_dim0_stride
  g1[6] = 0;
  g1[7] = 0;
  i32x4 z4 = {0, 0, 0, 0};
  i32x8 z8 = {0, 0, 0, 0, 0, 0, 0, 0};
  __builtin_amdgcn_tensor_load_to_lds(g0, g1, z4, z4, z8, 0);
}
#endif

// One stacked-RNN layer: out[t] = tanh(in[t] @ W + h_{t-1} @ U + b), all 50 steps.
__global__ __launch_bounds__(NTHR) void rnn_layer_kernel(
    const unsigned short* __restrict__ xin,   // [T][B][LK] bf16
    unsigned short* __restrict__ xout,        // [T][B][LK] bf16
    const float* __restrict__ W,              // [Kin][75] fp32
    const float* __restrict__ U,              // [75][75]  fp32
    const float* __restrict__ bias,           // [75]      fp32
    int Kin) {
  __shared__ __align__(16) unsigned short sWt[80 * LK];          // W^T [n][k]
  __shared__ __align__(16) unsigned short sUt[80 * LK];          // U^T [n][k]
  __shared__ __align__(16) unsigned short sH[2][MROWS * LK];     // h double buffer
  __shared__ __align__(16) unsigned short sX[2][MROWS * LK];     // x tile double buffer
  __shared__ float sBias[80];

  const int tid  = (int)threadIdx.x;
  const int wv   = tid >> 5;
  const int lane = tid & 31;
  const int nl   = lane & 15;
  const int hi8  = (lane >> 4) * 8;
  const int hi16 = (lane >> 4) * 16;
  const int rowbase = (int)blockIdx.x * MROWS;

  // Stage transposed, zero-padded weights into LDS (k>=Kin / n>=75 forced to 0,
  // which makes all padded K lanes contribute exactly zero).
  for (int i = tid; i < 80 * 96; i += NTHR) {
    const int n = i / 96, k = i - n * 96;
    unsigned short w = 0, u = 0;
    if (n < HIDC) {
      if (k < Kin)  w = f2bf(W[(size_t)k * HIDC + n]);
      if (k < HIDC) u = f2bf(U[(size_t)k * HIDC + n]);
    }
    sWt[n * LK + k] = w;
    sUt[n * LK + k] = u;
  }
  for (int i = tid; i < 2 * MROWS * LK; i += NTHR) (&sH[0][0])[i] = 0;  // h0 = 0
  for (int i = tid; i < 80; i += NTHR) sBias[i] = (i < HIDC) ? bias[i] : 0.0f;

  // Stage x tile for t = 0.
#if USE_TDM
  if (wv == 0) {
    tdm_tile_load(xin + (size_t)rowbase * LK, (void*)sX[0]);
    __builtin_amdgcn_s_wait_tensorcnt(0);
  }
#else
  {
    const uint4* g = (const uint4*)(xin + (size_t)rowbase * LK);
    uint4* d = (uint4*)sX[0];
    for (int i = tid; i < TILE_CHUNKS; i += NTHR) d[i] = g[i];
  }
#endif
  __syncthreads();

  // Loop-invariant B operands: this wave's 16 columns of W^T and U^T.
  bf16x16 BW[KCH], BU[KCH];
  {
    const unsigned short* wrow = &sWt[(wv * 16 + nl) * LK];
    const unsigned short* urow = &sUt[(wv * 16 + nl) * LK];
#pragma unroll
    for (int kc = 0; kc < KCH; ++kc) {
      const int ks = kc * 32 + hi8;
      BW[kc] = ld16(wrow + ks, wrow + ks + 16);
      BU[kc] = ld16(urow + ks, urow + ks + 16);
    }
  }
  const float cb = sBias[wv * 16 + nl];
  const f32x8 cinit = {cb, cb, cb, cb, cb, cb, cb, cb};
  const int col = wv * 16 + nl;

  for (int t = 0; t < SEQT; ++t) {
    const int cur = t & 1;
    const unsigned short* xb = sX[cur];
    const unsigned short* hr = sH[cur];
    unsigned short* hw = sH[cur ^ 1];

    // Kick off next x-tile transfer (overlaps with WMMA compute below).
#if USE_TDM
    if (wv == 0 && t + 1 < SEQT)
      tdm_tile_load(xin + ((size_t)(t + 1) * BATCH + rowbase) * LK,
                    (void*)sX[cur ^ 1]);
#else
    uint4 stg[3];
    if (t + 1 < SEQT) {
      const uint4* g = (const uint4*)(xin + ((size_t)(t + 1) * BATCH + rowbase) * LK);
#pragma unroll
      for (int c = 0; c < 3; ++c) {
        const int i = tid + c * NTHR;
        if (i < TILE_CHUNKS) stg[c] = g[i];
      }
      if (t + 2 < SEQT)
        __builtin_prefetch(xin + ((size_t)(t + 2) * BATCH + rowbase) * LK + tid * 8, 0, 1);
    }
#endif

#pragma unroll
    for (int mt = 0; mt < MROWS / 16; ++mt) {
      const unsigned short* ax = xb + (mt * 16 + nl) * LK;
      const unsigned short* ah = hr + (mt * 16 + nl) * LK;
      f32x8 acc = cinit;
#pragma unroll
      for (int kc = 0; kc < KCH; ++kc) {           // x_t @ W
        const int kb = kc * 32 + hi8;
        acc = wmma_bf16(ld16(ax + kb, ax + kb + 16), BW[kc], acc);
      }
#pragma unroll
      for (int kc = 0; kc < KCH; ++kc) {           // h_{t-1} @ U
        const int kb = kc * 32 + hi8;
        acc = wmma_bf16(ld16(ah + kb, ah + kb + 16), BU[kc], acc);
      }
      const int m0 = mt * 16 + hi8;
      unsigned short hb[8];
#pragma unroll
      for (int r = 0; r < 8; ++r) hb[r] = f2bf(TANHF(acc[r]));
#pragma unroll
      for (int r = 0; r < 8; ++r) hw[(m0 + r) * LK + col] = hb[r];
      unsigned short* go = xout + ((size_t)t * BATCH + rowbase + m0) * LK + col;
#pragma unroll
      for (int r = 0; r < 8; ++r) go[(size_t)r * LK] = hb[r];
    }

#if USE_TDM
    if (wv == 0 && t + 1 < SEQT) __builtin_amdgcn_s_wait_tensorcnt(0);
#else
    if (t + 1 < SEQT) {
      uint4* d = (uint4*)sX[cur ^ 1];
#pragma unroll
      for (int c = 0; c < 3; ++c) {
        const int i = tid + c * NTHR;
        if (i < TILE_CHUNKS) d[i] = stg[c];
      }
    }
#endif
    __syncthreads();
  }
}

// x [B][T][1] fp32 -> seq [T][B][LK] bf16 (column 0; rest pre-zeroed by memset).
__global__ void rnn_convert_kernel(const float* __restrict__ x,
                                   unsigned short* __restrict__ seq) {
  const int i = (int)(blockIdx.x * blockDim.x + threadIdx.x);
  if (i >= SEQT * BATCH) return;
  const int t = i / BATCH;
  const int b = i - t * BATCH;
  seq[(size_t)i * LK] = f2bf(x[(size_t)b * SEQT + t]);
}

// out[b] = h_last[b] . Wd + bd
__global__ void rnn_head_kernel(const unsigned short* __restrict__ seq,
                                const float* __restrict__ Wd,
                                const float* __restrict__ bd,
                                float* __restrict__ out) {
  const int b = (int)(blockIdx.x * blockDim.x + threadIdx.x);
  if (b >= BATCH) return;
  const unsigned short* row = seq + ((size_t)(SEQT - 1) * BATCH + b) * LK;
  float acc = bd[0];
  for (int n = 0; n < HIDC; ++n)
    acc += __uint_as_float((unsigned)row[n] << 16) * Wd[n];
  out[b] = acc;
}

extern "C" void kernel_launch(void* const* d_in, const int* in_sizes, int n_in,
                              void* d_out, int out_size, void* d_ws, size_t ws_size,
                              hipStream_t stream) {
  (void)in_sizes; (void)n_in; (void)out_size; (void)ws_size;
  const float* x  = (const float*)d_in[0];
  const float* W0 = (const float*)d_in[1];
  const float* U0 = (const float*)d_in[2];
  const float* b0 = (const float*)d_in[3];
  const float* Ws = (const float*)d_in[4];
  const float* Us = (const float*)d_in[5];
  const float* bs = (const float*)d_in[6];
  const float* Wd = (const float*)d_in[7];
  const float* bd = (const float*)d_in[8];
  float* out = (float*)d_out;

  const size_t seqElems = (size_t)SEQT * BATCH * LK;
  unsigned short* bufs[2] = {(unsigned short*)d_ws,
                             (unsigned short*)d_ws + seqElems};

  // Zero both sequence buffers so padded columns are exactly 0 (capturable).
  (void)hipMemsetAsync(d_ws, 0, 2 * seqElems * sizeof(unsigned short), stream);

  rnn_convert_kernel<<<(SEQT * BATCH + 255) / 256, 256, 0, stream>>>(x, bufs[0]);

  // Layer 0 (input dim 1), then layers 1..9 (75 -> 75), ping-ponging buffers.
  rnn_layer_kernel<<<BATCH / MROWS, NTHR, 0, stream>>>(bufs[0], bufs[1],
                                                       W0, U0, b0, 1);
  int cur = 1;
  for (int i = 0; i < 9; ++i) {
    rnn_layer_kernel<<<BATCH / MROWS, NTHR, 0, stream>>>(
        bufs[cur], bufs[cur ^ 1],
        Ws + (size_t)i * HIDC * HIDC,
        Us + (size_t)i * HIDC * HIDC,
        bs + (size_t)i * HIDC, HIDC);
    cur ^= 1;
  }

  rnn_head_kernel<<<(BATCH + 255) / 256, 256, 0, stream>>>(bufs[cur], Wd, bd, out);
}
